// GNN_59219009077824
// MI455X (gfx1250) — matrix operand order repaired and checked by reading, not compiled, not run on hardware
//
#include <hip/hip_runtime.h>
#include <hip/hip_bf16.h>

// ---------------------------------------------------------------------------
// Fused GCN (2x GCNConv + masked mean-pool) for B=1024, N=256, F=6, H=128, O=64.
//
// The normalized adjacency is rank-3 + diagonal:
//   (An M)_i = dinv_i * ( c_i ? S1 : Sc + dinv_i * M_i )
//   S1 = sum_j dinv_j M_j ,  Sc = sum_j c_j dinv_j M_j
//   deg_i = c_i ? Nv : Nc+1   (valid i), dinv_i = deg_i^{-1/2}
// so only the dense h1@W2 GEMM is O(N*H*O) -> v_wmma_f32_16x16x32_bf16.
// GEMM schedule: per wave, B fragments (nt = wave&3) are loaded once; per
// 16x16 tile all 8 A-fragment b128 loads are issued before the 4-WMMA chain
// so DS waits can be partial and LDS latency overlaps WMMA execution.
// ---------------------------------------------------------------------------

typedef __attribute__((ext_vector_type(16))) __bf16 v16bf;
typedef __attribute__((ext_vector_type(8)))  __bf16 v8bf;
typedef __attribute__((ext_vector_type(4)))  __bf16 v4bf;
typedef __attribute__((ext_vector_type(8)))  float  v8f;

#define N_NODES 256
#define F_IN    6
#define H_DIM   128
#define O_DIM   64
#define A_STRIDE 136   // padded bf16 row stride (272B = 68 words) -> conflict-free b128 frag loads

// ---- LDS layout (dynamic shared), all sections 16B aligned ----
constexpr size_t OFF_A    = 0;                                     // h1 / xW1, bf16 [256][136]
constexpr size_t OFF_W2T  = OFF_A    + (size_t)N_NODES*A_STRIDE*2; // W2^T bf16 [64][136]
constexpr size_t OFF_M2   = OFF_W2T  + (size_t)O_DIM*A_STRIDE*2;   // h1@W2 f32 [256][64]
constexpr size_t OFF_W1   = OFF_M2   + (size_t)N_NODES*O_DIM*4;
constexpr size_t OFF_B1   = OFF_W1   + (size_t)F_IN*H_DIM*4;
constexpr size_t OFF_B2   = OFF_B1   + (size_t)H_DIM*4;
constexpr size_t OFF_S1H  = OFF_B2   + (size_t)O_DIM*4;
constexpr size_t OFF_SCH  = OFF_S1H  + (size_t)H_DIM*4;
constexpr size_t OFF_S1O  = OFF_SCH  + (size_t)H_DIM*4;
constexpr size_t OFF_SCO  = OFF_S1O  + (size_t)O_DIM*4;
constexpr size_t OFF_DINV = OFF_SCO  + (size_t)O_DIM*4;
constexpr size_t OFF_WC   = OFF_DINV + (size_t)N_NODES*4;
constexpr size_t OFF_PART = OFF_WC   + (size_t)N_NODES*4;
constexpr size_t OFF_CNT  = OFF_PART + (size_t)4*O_DIM*4;
constexpr size_t SMEM_TOTAL = OFF_CNT + 16;

__global__ __launch_bounds__(256) void gcn_fused_kernel(
    const float* __restrict__ x,  const float* __restrict__ W1,
    const float* __restrict__ b1, const float* __restrict__ W2,
    const float* __restrict__ b2, float* __restrict__ out)
{
  extern __shared__ char smem[];
  __bf16* a_lds = (__bf16*)(smem + OFF_A);
  __bf16* w2t   = (__bf16*)(smem + OFF_W2T);
  float*  m2s   = (float*)(smem + OFF_M2);
  float*  w1s   = (float*)(smem + OFF_W1);
  float*  b1s   = (float*)(smem + OFF_B1);
  float*  b2s   = (float*)(smem + OFF_B2);
  float*  s1h   = (float*)(smem + OFF_S1H);
  float*  sch   = (float*)(smem + OFF_SCH);
  float*  s1o   = (float*)(smem + OFF_S1O);
  float*  sco   = (float*)(smem + OFF_SCO);
  float*  dinv  = (float*)(smem + OFF_DINV);
  float*  wC    = (float*)(smem + OFF_WC);
  float*  part  = (float*)(smem + OFF_PART);
  int*    cnts  = (int*)  (smem + OFF_CNT);

  const int b    = blockIdx.x;
  const int t    = threadIdx.x;         // 256 threads; thread t owns node t
  const int lane = t & 31;
  const int wave = t >> 5;              // 8 wave32s

  if (t == 0) { cnts[0] = 0; cnts[1] = 0; }
  __syncthreads();

  // ---- Stage weights: W1/b1/b2 (f32), W2 transposed -> bf16 [o][h] ----
  for (int i = t; i < F_IN*H_DIM; i += 256) w1s[i] = W1[i];
  for (int i = t; i < H_DIM;      i += 256) b1s[i] = b1[i];
  for (int i = t; i < O_DIM;      i += 256) b2s[i] = b2[i];
  for (int i = t; i < H_DIM*O_DIM; i += 256) {
    int h = i >> 6, o = i & 63;
    w2t[o*A_STRIDE + h] = (__bf16)W2[i];
  }

  // ---- Node flags + counts ----
  float xv[F_IN];
  const float* xr = x + ((size_t)b*N_NODES + t)*F_IN;
  float asum = 0.f;
  #pragma unroll
  for (int f = 0; f < F_IN; ++f) { xv[f] = xr[f]; asum += fabsf(xv[f]); }
  const bool valid = (asum != 0.f);
  const bool cond  = valid && ((xv[3] != 0.f) || (xv[4] != 0.f));
  if (valid) atomicAdd(&cnts[0], 1);
  if (cond)  atomicAdd(&cnts[1], 1);
  __syncthreads();

  const int Nv = cnts[0], Nc = cnts[1];
  float di = 0.f;
  if (valid) di = rsqrtf((float)(cond ? Nv : (Nc + 1)));
  dinv[t] = di;
  wC[t]   = cond ? di : 0.f;

  // ---- M1 = x @ W1 (K=6, VALU) -> bf16 in a_lds; float4 W1 reads, 8B stores ----
  for (int h4 = 0; h4 < H_DIM/4; ++h4) {
    float sx = 0.f, sy = 0.f, sz = 0.f, sw = 0.f;
    #pragma unroll
    for (int f = 0; f < F_IN; ++f) {
      float4 w = *(const float4*)&w1s[f*H_DIM + h4*4];
      sx += xv[f]*w.x; sy += xv[f]*w.y; sz += xv[f]*w.z; sw += xv[f]*w.w;
    }
    v4bf p;
    p[0] = (__bf16)sx; p[1] = (__bf16)sy; p[2] = (__bf16)sz; p[3] = (__bf16)sw;
    *(v4bf*)&a_lds[t*A_STRIDE + h4*4] = p;
  }
  __syncthreads();

  // ---- Rank-3 sums for layer 1: S1[h] (t<128), Sc[h] (t>=128) ----
  {
    const int  h      = t & (H_DIM - 1);
    const bool second = (t >= H_DIM);
    const float* w = second ? wC : dinv;
    float s = 0.f;
    for (int i = 0; i < N_NODES; ++i)
      s += w[i] * (float)a_lds[i*A_STRIDE + h];
    (second ? sch : s1h)[h] = s;
  }
  __syncthreads();

  // ---- h1 = relu(An*M1 + b1), in place (bf16), 4 lanes of h at a time ----
  for (int h4 = 0; h4 < H_DIM/4; ++h4) {
    v4bf p = *(const v4bf*)&a_lds[t*A_STRIDE + h4*4];
    v4bf q;
    #pragma unroll
    for (int j = 0; j < 4; ++j) {
      int h = h4*4 + j;
      float v = 0.f;
      if (valid) {
        float m   = (float)p[j];
        float agg = cond ? s1h[h] : (sch[h] + di*m);
        v = di*agg + b1s[h];
        v = v > 0.f ? v : 0.f;
      }
      q[j] = (__bf16)v;
    }
    *(v4bf*)&a_lds[t*A_STRIDE + h4*4] = q;
  }
  __syncthreads();

  // ---- M2 = h1 @ W2 via WMMA bf16 ----
  // Wave w: fixed N-tile nt = w&3, M-tiles mt = (w>>2) + 2q, q=0..7.
  {
    const int nt  = wave & 3;
    const int col = nt*16 + (lane & 15);
    const int blo = (lane < 16) ? 0 : 16;   // B K-halves: 0..15 | 16..31
    const int alo = (lane < 16) ? 0 : 16;   // A K-halves: 0..15 | 16..31

    // B fragments for all 4 K-steps: loaded once per wave.
    v16bf bfrag[4];
    #pragma unroll
    for (int kt = 0; kt < 4; ++kt) {
      v8bf bl = *(const v8bf*)&w2t[col*A_STRIDE + kt*32 + blo];
      v8bf bh = *(const v8bf*)&w2t[col*A_STRIDE + kt*32 + blo + 8];
      bfrag[kt] = __builtin_shufflevector(bl, bh,
          0,1,2,3,4,5,6,7,8,9,10,11,12,13,14,15);
    }

    for (int q = 0; q < 8; ++q) {
      const int mt  = (wave >> 2) + 2*q;
      const int row = mt*16 + (lane & 15);
      const __bf16* arow = &a_lds[row*A_STRIDE];

      // Issue all 8 independent A-fragment b128 loads up front -> partial waits.
      v8bf af[8];
      #pragma unroll
      for (int kt = 0; kt < 4; ++kt) {
        af[2*kt]   = *(const v8bf*)&arow[kt*32 + alo];
        af[2*kt+1] = *(const v8bf*)&arow[kt*32 + alo + 8];
      }

      v8f acc = {};
      #pragma unroll
      for (int kt = 0; kt < 4; ++kt) {
        v16bf afrag = __builtin_shufflevector(af[2*kt], af[2*kt+1],
            0,1,2,3,4,5,6,7,8,9,10,11,12,13,14,15);
        acc = __builtin_amdgcn_wmma_f32_16x16x32_bf16(
            false, afrag, false, bfrag[kt], (short)0, acc, false, false);
      }

      const int r0 = mt*16 + ((lane < 16) ? 0 : 8);
      #pragma unroll
      for (int r = 0; r < 8; ++r)
        m2s[(r0 + r)*O_DIM + col] = acc[r];
    }
  }
  __syncthreads();

  // ---- Rank-3 sums for layer 2: S1'[o] (t<64), Sc'[o] (64<=t<128) ----
  if (t < 2*O_DIM) {
    const int  o      = t & (O_DIM - 1);
    const bool second = (t >= O_DIM);
    const float* w = second ? wC : dinv;
    float s = 0.f;
    for (int i = 0; i < N_NODES; ++i)
      s += w[i] * m2s[i*O_DIM + o];
    (second ? sco : s1o)[o] = s;
  }
  __syncthreads();

  // ---- h2 + masked mean pool ----
  {
    const int o = t & 63, g = t >> 6;
    float s = 0.f;
    for (int i = g*64; i < (g+1)*64; ++i) {
      float dj = dinv[i];
      if (dj > 0.f) {                      // valid node
        bool  cj  = (wC[i] > 0.f);
        float agg = cj ? s1o[o] : (sco[o] + dj * m2s[i*O_DIM + o]);
        s += dj*agg + b2s[o];
      }
    }
    part[g*64 + o] = s;
  }
  __syncthreads();
  if (t < O_DIM) {
    float s = part[t] + part[64 + t] + part[128 + t] + part[192 + t];
    out[(size_t)b*O_DIM + t] = (Nv > 0) ? (s / (float)Nv) : 0.f;
  }
}

extern "C" void kernel_launch(void* const* d_in, const int* in_sizes, int n_in,
                              void* d_out, int out_size, void* d_ws, size_t ws_size,
                              hipStream_t stream) {
  const float* x  = (const float*)d_in[0];
  const float* W1 = (const float*)d_in[1];
  const float* b1 = (const float*)d_in[2];
  const float* W2 = (const float*)d_in[3];
  const float* b2 = (const float*)d_in[4];
  float* out = (float*)d_out;
  const int B = in_sizes[0] / (N_NODES * F_IN);   // 1024
  gcn_fused_kernel<<<B, 256, SMEM_TOTAL, stream>>>(x, W1, b1, W2, b2, out);
}
